// MultiHeadExternalAttention_11742440587784
// MI455X (gfx1250) — matrix operand order repaired and checked by reading, not compiled, not run on hardware
//
#include <hip/hip_runtime.h>

typedef __attribute__((ext_vector_type(16))) _Float16 v16h;
typedef __attribute__((ext_vector_type(4)))  _Float16 v4h;
typedef __attribute__((ext_vector_type(8)))  float    v8f;
typedef __attribute__((ext_vector_type(4)))  float    v4f;

#define NSPAT 4096   // H*W
#define LDH   80     // padded LDS row stride in halves (160B rows, 16B-aligned frags)

__device__ __forceinline__ v8f wmma_f16(v16h a, v16h b, v8f c) {
  // D = A(16x32 f16) * B(32x16 f16) + C(16x16 f32)
  return __builtin_amdgcn_wmma_f32_16x16x32_f16(
      false, a, false, b, (short)0, c, false, false);
}

// xor-lane swizzles (group-of-32 mode: offset = xor<<10 | 0x1F)
template <int PAT>
__device__ __forceinline__ float swz(float v) {
  return __int_as_float(__builtin_amdgcn_ds_swizzle(__float_as_int(v), PAT));
}
#define SWZ_X1  0x041F
#define SWZ_X2  0x081F
#define SWZ_X4  0x101F
#define SWZ_X8  0x201F
#define SWZ_X16 0x401F

__global__ __launch_bounds__(256)
void mea_fused_kernel(const float* __restrict__ q,
                      const float* __restrict__ mkw,   // [64 s][64 d]
                      const float* __restrict__ mvw,   // [64 d][64 s]
                      float* __restrict__ out) {
  __shared__ _Float16 s_mk[64 * LDH];    // mk_w f16, [s][d]
  __shared__ _Float16 s_mv[64 * LDH];    // mv_w f16, [d][s]
  __shared__ _Float16 s_x[8][16 * LDH];  // per-wave tile: x [n][d] f16, reused for probs [n][s]
  __shared__ float s_m[64 * 8];          // per-(s,wave) running max
  __shared__ float s_z[64 * 8];          // per-(s,wave) running sumexp
  __shared__ float s_M[64];              // final column max per s
  __shared__ float s_Zi[64];             // final 1/sumexp per s

  const int tid  = threadIdx.x;
  const int lane = tid & 31;
  const int w    = tid >> 5;                           // wave 0..7
  const long long qbase = (long long)blockIdx.x * 64 * NSPAT;  // (b*512 + h*64) rows
  const float* qp = q + qbase;
  float*       op = out + qbase;

  // Stage weights (f32 -> f16) and init softmax stat slots
  for (int i = tid; i < 64 * 64; i += 256) {
    int r = i >> 6, c = i & 63;
    s_mk[r * LDH + c] = (_Float16)mkw[i];
    s_mv[r * LDH + c] = (_Float16)mvw[i];
  }
  for (int i = tid; i < 64 * 8; i += 256) { s_m[i] = -3.0e38f; s_z[i] = 0.0f; }
  __syncthreads();

  const int  row   = lane & 15;
  const bool lo    = lane < 16;
  const int  sel8  = lo ? 0 : 8;     // D-row (n) offset: 8 * (lane / 16)
  const int  sel16 = lo ? 0 : 16;    // B-frag K sub-select

  // lane-private micro-tile: 4 rows (dd..dd+3) x 4 cols (n4..n4+3), two chunks
  // chunk = lane + it*32 ; n4 = (chunk&3)*4 ; dd = (chunk>>2)*4
  auto load_tile = [&](int n0, v4f (&R)[8], bool pf) {
#pragma unroll
    for (int it = 0; it < 2; ++it) {
      const int chunk = lane + it * 32;
      const int n4 = (chunk & 3) * 4;
      const int dd = (chunk >> 2) * 4;
      const float* g = qp + dd * NSPAT + n0 + n4;
#pragma unroll
      for (int j = 0; j < 4; ++j) {
        R[it * 4 + j] = *(const v4f*)(g + j * NSPAT);
        if (pf) __builtin_prefetch(g + j * NSPAT + 128, 0, 3);  // next-next tile
      }
    }
  };
  auto store_tile = [&](v4f (&R)[8]) {  // f32 regs -> f16 LDS [n][d]
    _Float16* xt = s_x[w];
#pragma unroll
    for (int it = 0; it < 2; ++it) {
      const int chunk = lane + it * 32;
      const int n4 = (chunk & 3) * 4;
      const int dd = (chunk >> 2) * 4;
#pragma unroll
      for (int j = 0; j < 4; ++j) {
        v4h h = { (_Float16)R[it * 4 + 0][j], (_Float16)R[it * 4 + 1][j],
                  (_Float16)R[it * 4 + 2][j], (_Float16)R[it * 4 + 3][j] };
        *(v4h*)(xt + (n4 + j) * LDH + dd) = h;   // 8B-aligned ds_store_b64
      }
    }
  };
  auto make_afrags = [&](const _Float16* xt, v16h& a0, v16h& a1) {
#pragma unroll
    for (int i = 0; i < 8; ++i) {
      a0[i]     = xt[row * LDH +      sel16 + i];
      a0[i + 8] = xt[row * LDH +  8 + sel16 + i];
      a1[i]     = xt[row * LDH + 32 + sel16 + i];
      a1[i + 8] = xt[row * LDH + 40 + sel16 + i];
    }
  };

  // ================= PASS 1: spatial-softmax column stats =================
  {
    v4f R0[8], R1[8];
    load_tile(w * 16, R0, true);
    for (int nb = w; nb < NSPAT / 16; nb += 8) {
      store_tile(R0);
      if (nb + 8 < NSPAT / 16) load_tile((nb + 8) * 16, R1, true);  // overlap
      asm volatile("s_wait_dscnt 0" ::: "memory");

      _Float16* xt = s_x[w];
      v16h a0, a1;
      make_afrags(xt, a0, a1);

#pragma unroll
      for (int st = 0; st < 4; ++st) {
        const int s = st * 16 + row;
        v16h b0, b1;  // columns of mk^T == rows of mk (contiguous in d)
#pragma unroll
        for (int i = 0; i < 16; ++i) {
          b0[i] = s_mk[s * LDH +      sel16 + i];
          b1[i] = s_mk[s * LDH + 32 + sel16 + i];
        }
        v8f c = {};
        c = wmma_f16(a0, b0, c);
        c = wmma_f16(a1, b1, c);

        // lane holds column s, 8 n-values -> partial (max, sumexp)
        float m = c[0];
#pragma unroll
        for (int r = 1; r < 8; ++r) m = fmaxf(m, c[r]);
        float z = 0.0f;
#pragma unroll
        for (int r = 0; r < 8; ++r) z += __expf(c[r] - m);
        // merge with partner lane (same s, other 8 n) via ds_swizzle SWAPX16
        float mo = swz<SWZ_X16>(m);
        float zo = swz<SWZ_X16>(z);
        float mm = fmaxf(m, mo);
        float zz = z * __expf(m - mm) + zo * __expf(mo - mm);
        if (lo) {  // one lane per s updates this wave's private slot
          float ms = s_m[s * 8 + w];
          float zs = s_z[s * 8 + w];
          float mn = fmaxf(ms, mm);
          s_z[s * 8 + w] = zs * __expf(ms - mn) + zz * __expf(mm - mn);
          s_m[s * 8 + w] = mn;
        }
      }
#pragma unroll
      for (int i = 0; i < 8; ++i) R0[i] = R1[i];
    }
  }
  __syncthreads();
  if (tid < 64) {  // fold 8 wave slots per s
    float M = -3.0e38f, Z = 0.0f;
    for (int i = 0; i < 8; ++i) {
      float m = s_m[tid * 8 + i], z = s_z[tid * 8 + i];
      float mn = fmaxf(M, m);
      Z = Z * __expf(M - mn) + z * __expf(m - mn);
      M = mn;
    }
    s_M[tid]  = M;
    s_Zi[tid] = 1.0f / Z;
  }
  __syncthreads();

  float Ms[4], Zi[4];
#pragma unroll
  for (int st = 0; st < 4; ++st) {
    Ms[st] = s_M[st * 16 + row];
    Zi[st] = s_Zi[st * 16 + row];
  }

  // ====== PASS 2: recompute logits, normalize, GEMM2, residual store ======
  {
    v4f R0[8], R1[8];
    load_tile(w * 16, R0, false);   // hot in L2 from pass 1
    for (int nb = w; nb < NSPAT / 16; nb += 8) {
      const int n0 = nb * 16;
      store_tile(R0);
      if (nb + 8 < NSPAT / 16) load_tile((nb + 8) * 16, R1, false);
      asm volatile("s_wait_dscnt 0" ::: "memory");

      _Float16* xt = s_x[w];
      v16h a0, a1;
      make_afrags(xt, a0, a1);

      float f[4][8];
      float rs[8];
#pragma unroll
      for (int r = 0; r < 8; ++r) rs[r] = 0.0f;

#pragma unroll
      for (int st = 0; st < 4; ++st) {
        const int s = st * 16 + row;
        v16h b0, b1;
#pragma unroll
        for (int i = 0; i < 16; ++i) {
          b0[i] = s_mk[s * LDH +      sel16 + i];
          b1[i] = s_mk[s * LDH + 32 + sel16 + i];
        }
        v8f c = {};
        c = wmma_f16(a0, b0, c);
        c = wmma_f16(a1, b1, c);
#pragma unroll
        for (int r = 0; r < 8; ++r) {
          float e = __expf(c[r] - Ms[st]) * Zi[st];  // spatial softmax
          f[st][r] = e;
          rs[r] += e;                                 // L1 row-sum over S
        }
      }
      // reduce row sums across the 16 lanes of each half (same n set)
#pragma unroll
      for (int r = 0; r < 8; ++r) rs[r] += swz<SWZ_X1>(rs[r]);
#pragma unroll
      for (int r = 0; r < 8; ++r) rs[r] += swz<SWZ_X2>(rs[r]);
#pragma unroll
      for (int r = 0; r < 8; ++r) rs[r] += swz<SWZ_X4>(rs[r]);
#pragma unroll
      for (int r = 0; r < 8; ++r) rs[r] += swz<SWZ_X8>(rs[r]);
#pragma unroll
      for (int r = 0; r < 8; ++r) rs[r] = 1.0f / rs[r];

      // x tile is dead (a0/a1 in regs): overwrite with normalized probs [n][s] f16
      _Float16* ft = xt;
#pragma unroll
      for (int st = 0; st < 4; ++st) {
        const int s = st * 16 + row;
#pragma unroll
        for (int r = 0; r < 8; ++r)
          ft[(r + sel8) * LDH + s] = (_Float16)(f[st][r] * rs[r]);
      }
      asm volatile("s_wait_dscnt 0" ::: "memory");

      v16h p0, p1;
      make_afrags(ft, p0, p1);
#pragma unroll
      for (int dt = 0; dt < 4; ++dt) {
        const int di = dt * 16 + row;
        v16h b0, b1;  // columns of mv^T == rows of mv (contiguous in s)
#pragma unroll
        for (int i = 0; i < 16; ++i) {
          b0[i] = s_mv[di * LDH +      sel16 + i];
          b1[i] = s_mv[di * LDH + 32 + sel16 + i];
        }
        v8f c = {};
        c = wmma_f16(p0, b0, c);
        c = wmma_f16(p1, b1, c);
        // lane holds channel di, n = n0 + sel8 + 0..7 : 32B contiguous, aligned
        const long long off = (long long)di * NSPAT + n0 + sel8;
        const v4f q0 = *(const v4f*)(qp + off);
        const v4f q1 = *(const v4f*)(qp + off + 4);
        v4f o0 = { c[0] + q0[0], c[1] + q0[1], c[2] + q0[2], c[3] + q0[3] };
        v4f o1 = { c[4] + q1[0], c[5] + q1[1], c[6] + q1[2], c[7] + q1[3] };
        *(v4f*)(op + off)     = o0;
        *(v4f*)(op + off + 4) = o1;
      }
#pragma unroll
      for (int i = 0; i < 8; ++i) R0[i] = R1[i];
    }
  }
}

extern "C" void kernel_launch(void* const* d_in, const int* in_sizes, int n_in,
                              void* d_out, int out_size, void* d_ws, size_t ws_size,
                              hipStream_t stream) {
  (void)in_sizes; (void)n_in; (void)out_size; (void)d_ws; (void)ws_size;
  const float* q   = (const float*)d_in[0];
  const float* mkw = (const float*)d_in[1];
  const float* mvw = (const float*)d_in[2];
  float* out = (float*)d_out;
  // one workgroup per (b, head): 32 * 8 = 256
  mea_fused_kernel<<<dim3(256), dim3(256), 0, stream>>>(q, mkw, mvw, out);
}
